// LocalCausalTransformerBlock_91293824843838
// MI455X (gfx1250) — compile-verified, matrix operand order checked by reading, not run on hardware
//
#include <hip/hip_runtime.h>
#include <hip/hip_bf16.h>
#include <math.h>

// ---------------------------------------------------------------------------
// Types for WMMA fragments (gfx1250, wave32)
// ---------------------------------------------------------------------------
typedef _Float16 half_t;
typedef __attribute__((ext_vector_type(16))) _Float16 v16h;
typedef __attribute__((ext_vector_type(8)))  _Float16 v8h;
typedef __attribute__((ext_vector_type(8)))  float    v8f;

#define TOKENS   4096      // B*N = 2*2048
#define SEQ      2048
#define DIMC     1024
#define NHEADS   16
#define HDIM     64
#define HIDDENC  4096
#define LN_EPS   1e-5f

// ---------------------------------------------------------------------------
// gfx1250 async-to-LDS copy (16B per lane, tracked by ASYNCcnt)
// ---------------------------------------------------------------------------
__device__ __forceinline__ void async_lds_copy16(unsigned lds_off, const void* gaddr) {
    asm volatile("global_load_async_to_lds_b128 %0, %1, off"
                 :: "v"(lds_off), "v"(gaddr) : "memory");
}
__device__ __forceinline__ void wait_async0() {
#if __has_builtin(__builtin_amdgcn_s_wait_asynccnt)
    __builtin_amdgcn_s_wait_asynccnt(0);
#else
    asm volatile("s_wait_asynccnt 0" ::: "memory");
#endif
}

// ---------------------------------------------------------------------------
// fp32 -> f16 conversion (weights)
// ---------------------------------------------------------------------------
__global__ void cvt_f32_to_f16(const float* __restrict__ in, half_t* __restrict__ out, int n) {
    int i = blockIdx.x * blockDim.x + threadIdx.x;
    if (i < n) out[i] = (half_t)in[i];
}

// ---------------------------------------------------------------------------
// LayerNorm over last dim (C=1024), one block (256 threads) per row -> f16.
// ---------------------------------------------------------------------------
__global__ __launch_bounds__(256)
void ln_kernel(const float* __restrict__ x, const float* __restrict__ g,
               const float* __restrict__ b, half_t* __restrict__ out) {
    const int C = DIMC;
    int row = blockIdx.x;
    int tid = threadIdx.x;
    const float4* xr = (const float4*)(x + (size_t)row * C);
    float4 v = xr[tid];
    float s  = v.x + v.y + v.z + v.w;
    float s2 = v.x*v.x + v.y*v.y + v.z*v.z + v.w*v.w;

    __shared__ float rs[256];
    __shared__ float rs2[256];
    rs[tid] = s; rs2[tid] = s2;
    __syncthreads();
    for (int o = 128; o > 0; o >>= 1) {
        if (tid < o) { rs[tid] += rs[tid + o]; rs2[tid] += rs2[tid + o]; }
        __syncthreads();
    }
    float mu  = rs[0] * (1.0f / C);
    float var = rs2[0] * (1.0f / C) - mu * mu;
    float inv = rsqrtf(var + LN_EPS);

    float4 gg = ((const float4*)g)[tid];
    float4 bb = ((const float4*)b)[tid];
    half_t* orow = out + (size_t)row * C;
    int i0 = tid * 4;
    orow[i0 + 0] = (half_t)((v.x - mu) * inv * gg.x + bb.x);
    orow[i0 + 1] = (half_t)((v.y - mu) * inv * gg.y + bb.y);
    orow[i0 + 2] = (half_t)((v.z - mu) * inv * gg.z + bb.z);
    orow[i0 + 3] = (half_t)((v.w - mu) * inv * gg.w + bb.w);
}

// ---------------------------------------------------------------------------
// Windowed (WINDOW=3) causal attention; 16 lanes per (row, head).
// ---------------------------------------------------------------------------
__global__ __launch_bounds__(256)
void attn_kernel(const float* __restrict__ qkv, half_t* __restrict__ out) {
    int t      = blockIdx.x * blockDim.x + threadIdx.x;
    int lane16 = t & 15;
    int hd     = t >> 4;
    int head   = hd & (NHEADS - 1);
    int row    = hd >> 4;
    int n      = row & (SEQ - 1);

    const float* base = qkv + (size_t)row * (3 * DIMC);
    int d0 = lane16 * 4;
    float4 q4 = *(const float4*)(base + head * HDIM + d0);

    const float scale = 0.125f;   // 1/sqrt(64)
    float sc[3];
    #pragma unroll
    for (int jj = 0; jj < 3; ++jj) {
        int j = n - 2 + jj;
        if (j >= 0) {
            const float* kb = qkv + (size_t)(row + jj - 2) * (3 * DIMC) + DIMC + head * HDIM + d0;
            float4 k4 = *(const float4*)kb;
            float p = q4.x * k4.x + q4.y * k4.y + q4.z * k4.z + q4.w * k4.w;
            #pragma unroll
            for (int m = 8; m >= 1; m >>= 1) p += __shfl_xor(p, m, 16);
            sc[jj] = p * scale;
        } else {
            sc[jj] = -1e30f;
        }
    }
    float mx = fmaxf(sc[0], fmaxf(sc[1], sc[2]));
    float e0 = __expf(sc[0] - mx), e1 = __expf(sc[1] - mx), e2 = __expf(sc[2] - mx);
    float inv = 1.0f / (e0 + e1 + e2);
    float w[3] = { e0 * inv, e1 * inv, e2 * inv };

    float4 o = make_float4(0.f, 0.f, 0.f, 0.f);
    #pragma unroll
    for (int jj = 0; jj < 3; ++jj) {
        int j = n - 2 + jj;
        if (j >= 0) {
            const float* vb = qkv + (size_t)(row + jj - 2) * (3 * DIMC) + 2 * DIMC + head * HDIM + d0;
            float4 v4 = *(const float4*)vb;
            o.x += w[jj] * v4.x; o.y += w[jj] * v4.y;
            o.z += w[jj] * v4.z; o.w += w[jj] * v4.w;
        }
    }
    half_t* orow = out + (size_t)row * DIMC + head * HDIM + d0;
    orow[0] = (half_t)o.x; orow[1] = (half_t)o.y;
    orow[2] = (half_t)o.z; orow[3] = (half_t)o.w;
}

// ---------------------------------------------------------------------------
// WMMA GEMM: C[M,N] = A[M,K](f16) @ W[K,N](f16) + bias; epilogues:
//   EPI 0: fp32 (bias) ; EPI 1: fp32 (bias+residual) ; EPI 2: f16 (bias+GELU)
// 256 threads = 8 waves; tile 128x128, BK=64 (2 WMMA K-steps / stage).
// Double-buffered: A staged via async-to-LDS, B via register pipeline with
// transpose store (Bs[n][k]) so both fragments are 2x ds_load_b128.
// ---------------------------------------------------------------------------
#define BM 128
#define BN 128
#define BK 64
#define LDA 72   // padded stride (halves): 36 banks -> conflict-free b128 reads
#define LDB 72

template<int EPI, int N, int K>
__global__ __launch_bounds__(256)
void gemm_wmma(const half_t* __restrict__ A, const half_t* __restrict__ Wt,
               const float* __restrict__ bias, const float* __restrict__ res,
               float* __restrict__ outF, half_t* __restrict__ outH) {
    __shared__ half_t As[2][BM * LDA];   // As[buf][m][k]
    __shared__ half_t Bs[2][BN * LDB];   // Bs[buf][n][k] (transposed stage)

    const int tid  = threadIdx.x;
    const int lane = tid & 31;
    const int wave = tid >> 5;
    const int wr   = wave & 1;    // 2 row-groups of 64
    const int wc   = wave >> 1;   // 4 col-groups of 32
    const int bm   = blockIdx.y * BM;
    const int bn   = blockIdx.x * BN;

    const int r16 = lane & 15;
    const int kb  = (lane >> 4) * 8;  // lanes 0-15: K 0-7/16-23; 16-31: K 8-15/24-31

    v8f acc[4][2];
    #pragma unroll
    for (int mt = 0; mt < 4; ++mt)
        #pragma unroll
        for (int nt = 0; nt < 2; ++nt)
            acc[mt][nt] = (v8f)(0.0f);

    // ---- staging helpers (each thread owns 4 16B chunks of each tile) ----
    auto issueA = [&](int kt, int buf) {               // A tile: 128 x 64 halves
        #pragma unroll
        for (int j = 0; j < 4; ++j) {
            int c  = tid + j * 256;
            int r  = c >> 3;              // 8 chunks per row
            int kc = (c & 7) * 8;
            unsigned lds = (unsigned)(uintptr_t)&As[buf][r * LDA + kc];
            const half_t* g = A + (size_t)(bm + r) * K + kt * BK + kc;
            async_lds_copy16(lds, g);
        }
    };
    auto loadB = [&](int kt, v8h breg[4]) {            // B tile: 64 x 128 halves
        #pragma unroll
        for (int j = 0; j < 4; ++j) {
            int c  = tid + j * 256;
            int kr = c >> 4;              // 16 chunks per row
            int nc = (c & 15) * 8;
            breg[j] = *(const v8h*)(Wt + (size_t)(kt * BK + kr) * N + bn + nc);
        }
    };
    auto storeB = [&](int buf, v8h breg[4]) {          // transpose to Bs[n][k]
        #pragma unroll
        for (int j = 0; j < 4; ++j) {
            int c  = tid + j * 256;
            int kr = c >> 4;
            int nc = (c & 15) * 8;
            #pragma unroll
            for (int i = 0; i < 8; ++i)
                Bs[buf][(nc + i) * LDB + kr] = breg[j][i];
        }
    };

    const int NT = K / BK;
    v8h breg[4];
    issueA(0, 0);
    loadB(0, breg);

    for (int it = 0; it < NT; ++it) {
        const int buf = it & 1;
        storeB(buf, breg);        // consume pipeline regs (tile it)
        wait_async0();            // my async A copies for tile it complete
        __syncthreads();          // tile it fully staged across all waves

        if (it + 1 < NT) {        // overlap next tile's traffic with WMMAs
            issueA(it + 1, buf ^ 1);
            loadB(it + 1, breg);
        }

        #pragma unroll
        for (int s = 0; s < 2; ++s) {         // two WMMA K-steps per stage
            v16h bfrag[2];
            #pragma unroll
            for (int nt = 0; nt < 2; ++nt) {
                int n = wc * 32 + nt * 16 + r16;
                v8h lo = *(const v8h*)&Bs[buf][n * LDB + s * 32 + kb];
                v8h hi = *(const v8h*)&Bs[buf][n * LDB + s * 32 + kb + 16];
                bfrag[nt] = __builtin_shufflevector(lo, hi, 0,1,2,3,4,5,6,7,8,9,10,11,12,13,14,15);
            }
            #pragma unroll
            for (int mt = 0; mt < 4; ++mt) {
                int r = wr * 64 + mt * 16 + r16;
                v8h lo = *(const v8h*)&As[buf][r * LDA + s * 32 + kb];
                v8h hi = *(const v8h*)&As[buf][r * LDA + s * 32 + kb + 16];
                v16h afrag = __builtin_shufflevector(lo, hi, 0,1,2,3,4,5,6,7,8,9,10,11,12,13,14,15);
                #pragma unroll
                for (int nt = 0; nt < 2; ++nt) {
                    acc[mt][nt] = __builtin_amdgcn_wmma_f32_16x16x32_f16(
                        false, afrag, false, bfrag[nt], (short)0, acc[mt][nt], false, false);
                }
            }
        }
        // no second barrier needed: buf^1 is only overwritten after the
        // barrier that proves all waves finished reading it (one iter ago).
    }

    // ---- epilogue: C layout: N = lane&15; M = (lane>>4)*8 + v ----
    #pragma unroll
    for (int mt = 0; mt < 4; ++mt) {
        #pragma unroll
        for (int nt = 0; nt < 2; ++nt) {
            int n = bn + wc * 32 + nt * 16 + r16;
            float bv = bias[n];
            int rbase = bm + wr * 64 + mt * 16 + (lane >> 4) * 8;
            #pragma unroll
            for (int v = 0; v < 8; ++v) {
                int r = rbase + v;
                float val = acc[mt][nt][v] + bv;
                if (EPI == 1) {
                    val += res[(size_t)r * N + n];
                    outF[(size_t)r * N + n] = val;
                } else if (EPI == 2) {
                    float ge = 0.5f * val * (1.0f + erff(val * 0.70710678118654752f));
                    outH[(size_t)r * N + n] = (half_t)ge;
                } else {
                    outF[(size_t)r * N + n] = val;
                }
            }
        }
    }
}

// ---------------------------------------------------------------------------
// Host launcher
// ---------------------------------------------------------------------------
extern "C" void kernel_launch(void* const* d_in, const int* in_sizes, int n_in,
                              void* d_out, int out_size, void* d_ws, size_t ws_size,
                              hipStream_t stream) {
    const float* x      = (const float*)d_in[0];
    const float* qkv_w  = (const float*)d_in[1];
    const float* qkv_b  = (const float*)d_in[2];
    const float* proj_w = (const float*)d_in[3];
    const float* proj_b = (const float*)d_in[4];
    const float* g1     = (const float*)d_in[5];
    const float* b1     = (const float*)d_in[6];
    const float* g2     = (const float*)d_in[7];
    const float* b2     = (const float*)d_in[8];
    const float* fc1_w  = (const float*)d_in[9];
    const float* fc1_b  = (const float*)d_in[10];
    const float* fc2_w  = (const float*)d_in[11];
    const float* fc2_b  = (const float*)d_in[12];
    float* out = (float*)d_out;

    char* ws = (char*)d_ws;
    size_t off = 0;
    auto alloc = [&](size_t bytes) { size_t o = off; off += (bytes + 255) & ~(size_t)255; return o; };

    half_t* qkv_w16  = (half_t*)(ws + alloc((size_t)DIMC * 3 * DIMC * 2));
    half_t* proj_w16 = (half_t*)(ws + alloc((size_t)DIMC * DIMC * 2));
    half_t* fc1_w16  = (half_t*)(ws + alloc((size_t)DIMC * HIDDENC * 2));
    half_t* fc2_w16  = (half_t*)(ws + alloc((size_t)HIDDENC * DIMC * 2));
    half_t* hA16     = (half_t*)(ws + alloc((size_t)TOKENS * DIMC * 2));
    float*  qkv_f32  = (float*) (ws + alloc((size_t)TOKENS * 3 * DIMC * 4));
    half_t* attn16   = (half_t*)(ws + alloc((size_t)TOKENS * DIMC * 2));
    float*  x1_f32   = (float*) (ws + alloc((size_t)TOKENS * DIMC * 4));
    half_t* h2_16    = (half_t*)(ws + alloc((size_t)TOKENS * DIMC * 2));
    half_t* g16      = (half_t*)(ws + alloc((size_t)TOKENS * HIDDENC * 2));
    (void)ws_size; (void)in_sizes; (void)n_in; (void)out_size;

    // 1) weight conversions fp32 -> f16
    {
        int n;
        n = DIMC * 3 * DIMC;   cvt_f32_to_f16<<<(n + 255) / 256, 256, 0, stream>>>(qkv_w,  qkv_w16,  n);
        n = DIMC * DIMC;       cvt_f32_to_f16<<<(n + 255) / 256, 256, 0, stream>>>(proj_w, proj_w16, n);
        n = DIMC * HIDDENC;    cvt_f32_to_f16<<<(n + 255) / 256, 256, 0, stream>>>(fc1_w,  fc1_w16,  n);
        n = HIDDENC * DIMC;    cvt_f32_to_f16<<<(n + 255) / 256, 256, 0, stream>>>(fc2_w,  fc2_w16,  n);
    }

    // 2) LN1
    ln_kernel<<<TOKENS, 256, 0, stream>>>(x, g1, b1, hA16);

    // 3) QKV GEMM: [4096 x 3072] = hA16 @ qkv_w16 + qkv_b
    {
        dim3 grid((3 * DIMC) / BN, TOKENS / BM);
        gemm_wmma<0, 3 * DIMC, DIMC><<<grid, 256, 0, stream>>>(hA16, qkv_w16, qkv_b, nullptr,
                                                               qkv_f32, nullptr);
    }

    // 4) windowed attention -> attn16
    attn_kernel<<<(TOKENS * NHEADS * 16) / 256, 256, 0, stream>>>(qkv_f32, attn16);

    // 5) proj GEMM + residual: x1 = x + attn16 @ proj_w16 + proj_b
    {
        dim3 grid(DIMC / BN, TOKENS / BM);
        gemm_wmma<1, DIMC, DIMC><<<grid, 256, 0, stream>>>(attn16, proj_w16, proj_b, x,
                                                           x1_f32, nullptr);
    }

    // 6) LN2
    ln_kernel<<<TOKENS, 256, 0, stream>>>(x1_f32, g2, b2, h2_16);

    // 7) FC1 GEMM + exact GELU -> g16 (f16)
    {
        dim3 grid(HIDDENC / BN, TOKENS / BM);
        gemm_wmma<2, HIDDENC, DIMC><<<grid, 256, 0, stream>>>(h2_16, fc1_w16, fc1_b, nullptr,
                                                              nullptr, g16);
    }

    // 8) FC2 GEMM + residual -> d_out (fp32)
    {
        dim3 grid(DIMC / BN, TOKENS / BM);
        gemm_wmma<1, DIMC, HIDDENC><<<grid, 256, 0, stream>>>(g16, fc2_w16, fc2_b, x1_f32,
                                                              out, nullptr);
    }
}
